// Bengio_33122787787245
// MI455X (gfx1250) — matrix operand order repaired and checked by reading, not compiled, not run on hardware
//
#include <hip/hip_runtime.h>
#include <hip/hip_bf16.h>
#include <math.h>

typedef __attribute__((ext_vector_type(16))) __bf16 v16bf;
typedef __attribute__((ext_vector_type(8)))  __bf16 v8bf;
typedef __attribute__((ext_vector_type(4)))  __bf16 v4bf;
typedef __attribute__((ext_vector_type(8)))  float  v8f;
typedef __attribute__((ext_vector_type(4)))  int    v4i;

#define BATCHN 4096
#define CTXN   8
#define EMBN   128
#define HIDN   1000
#define VOCABN 50257
#define XCOLS  2048      // padded x_cat width: [h 0..999 | embed 1000..2023 | zero 2024..2047]
#define KREAL2 2024
#define LDSTRD 40        // LDS tile row stride in bf16 elements (80B: 16B-aligned, bank-spread)

// ---- CDNA5 async global->LDS path (ASYNCcnt), guarded so we never lose compile ----
#if defined(__has_builtin)
#if __has_builtin(__builtin_amdgcn_global_load_async_to_lds_b128)
#define HAVE_ASYNC_LDS 1
#endif
#endif
#ifndef HAVE_ASYNC_LDS
#define HAVE_ASYNC_LDS 0
#endif

#if HAVE_ASYNC_LDS
#if __has_builtin(__builtin_amdgcn_s_wait_asynccnt)
#define ASYNC_WAIT0() __builtin_amdgcn_s_wait_asynccnt(0)
#else
#define ASYNC_WAIT0() asm volatile("s_wait_asynccnt 0x0" ::: "memory")
#endif
// builtin signature (from clang diagnostic): (v4i AS1*, v4i AS3*, imm int offset, imm int cpol)
typedef __attribute__((address_space(1))) v4i* gv4ip;
typedef __attribute__((address_space(3))) v4i* lv4ip;
#define GLD_ASYNC_B128(gp, lp) \
    __builtin_amdgcn_global_load_async_to_lds_b128((gv4ip)(void*)(gp), (lv4ip)(void*)(lp), 0, 0)
#else
#define ASYNC_WAIT0() ((void)0)
#endif

// ---------------------------------------------------------------------------
// Kernel A: embedding gather + bf16 hi/lo split into x[:, 1000..2047]
// ---------------------------------------------------------------------------
__global__ __launch_bounds__(256) void embed_split_kernel(
    const int* __restrict__ ctx, const float* __restrict__ We,
    const float* __restrict__ be,
    __bf16* __restrict__ xhi, __bf16* __restrict__ xlo)
{
    int b = blockIdx.x;
    __shared__ int toks[CTXN];
    if (threadIdx.x < CTXN) toks[threadIdx.x] = ctx[b * CTXN + threadIdx.x];
    __syncthreads();
    for (int j = threadIdx.x; j < (CTXN * EMBN + 24); j += 256) {
        float v = 0.0f;
        if (j < CTXN * EMBN) {
            int c = j >> 7, e = j & 127;
            v = We[(size_t)e * VOCABN + toks[c]] + be[e];
        }
        __bf16 hi = (__bf16)v;
        __bf16 lo = (__bf16)(v - (float)hi);
        size_t o = (size_t)b * XCOLS + HIDN + j;
        xhi[o] = hi;
        xlo[o] = lo;
    }
}

// ---------------------------------------------------------------------------
// Fragment loaders (CDNA5 ISA 7.12.2 layouts, 16-bit elements, wave32)
// ---------------------------------------------------------------------------
__device__ __forceinline__ v16bf cat8(v8bf a, v8bf b) {
    return __builtin_shufflevector(a, b, 0,1,2,3,4,5,6,7,8,9,10,11,12,13,14,15);
}
// A 16x32: lane m=lane&15, kh=lane>>4; VGPR0-3: K=8*kh+0..7, VGPR4-7: K=16+8*kh+0..7
__device__ __forceinline__ v16bf load_frag_a(const __bf16* tile, int lane) {
    int m = lane & 15, kh = lane >> 4;
    const __bf16* row = tile + m * LDSTRD + kh * 8;
    return cat8(*(const v8bf*)row, *(const v8bf*)(row + 16));
}
// B 32x16 staged as W-rows (n-major): lane n=lane&15, kh=lane>>4; VGPR0-3: K=8*kh+0..7, VGPR4-7: K=16+8*kh+0..7
__device__ __forceinline__ v16bf load_frag_b(const __bf16* tile, int lane) {
    int n = lane & 15, kh = lane >> 4;
    const __bf16* row = tile + n * LDSTRD + kh * 8;
    return cat8(*(const v8bf*)row, *(const v8bf*)(row + 16));
}

#define WMMA_BF16(A, B, C) \
    __builtin_amdgcn_wmma_f32_16x16x32_bf16(false, (A), false, (B), (short)0, (C), false, false)

// ---------------------------------------------------------------------------
// Pipeline stages
// ---------------------------------------------------------------------------
// A tiles (pre-split bf16 hi/lo in global): 128x32 each; 2 x b128 per thread per matrix.
__device__ __forceinline__ void stage_A(
    const __bf16* __restrict__ Ahi, const __bf16* __restrict__ Alo,
    int mBase, int aColOff, int kBase, int t,
    __bf16* dAhi, __bf16* dAlo)
{
#pragma unroll
    for (int it = 0; it < 2; ++it) {
        int chunk = t + (it << 8);           // 0..511
        int r  = chunk >> 2;
        int c8 = (chunk & 3) << 3;           // 0,8,16,24 (bf16 elems)
        size_t g = (size_t)(mBase + r) * XCOLS + aColOff + kBase + c8;
        __bf16* lhi = dAhi + r * LDSTRD + c8;
        __bf16* llo = dAlo + r * LDSTRD + c8;
#if HAVE_ASYNC_LDS
        GLD_ASYNC_B128(Ahi + g, lhi);
        GLD_ASYNC_B128(Alo + g, llo);
#else
        *(v8bf*)lhi = *(const v8bf*)(Ahi + g);
        *(v8bf*)llo = *(const v8bf*)(Alo + g);
#endif
    }
}

// B tile global->regs (fp32, bounds-guarded): 16 floats per thread.
__device__ __forceinline__ void load_B(
    const float* __restrict__ W, int ldw, int N, int Kreal,
    int nBase, int kBase, int t, float* __restrict__ wr)
{
    int r0 = t >> 3, cq = (t & 7) << 2;
#pragma unroll
    for (int p = 0; p < 4; ++p) {
        int r  = (p << 5) + r0;
        int n  = nBase + r;
        int kb = kBase + cq;
        float w0 = 0.f, w1 = 0.f, w2 = 0.f, w3 = 0.f;
        if (n < N) {
            const float* src = W + (size_t)n * ldw + kb;
            if (kb + 3 < Kreal) {
                float4 f = *(const float4*)src;
                w0 = f.x; w1 = f.y; w2 = f.z; w3 = f.w;
            } else {
                if (kb + 0 < Kreal) w0 = src[0];
                if (kb + 1 < Kreal) w1 = src[1];
                if (kb + 2 < Kreal) w2 = src[2];
                if (kb + 3 < Kreal) w3 = src[3];
            }
        }
        wr[p * 4 + 0] = w0; wr[p * 4 + 1] = w1;
        wr[p * 4 + 2] = w2; wr[p * 4 + 3] = w3;
    }
}

// B regs -> bf16 hi/lo split -> LDS.
__device__ __forceinline__ void store_B(
    const float* __restrict__ wr, int t, __bf16* dBhi, __bf16* dBlo)
{
    int r0 = t >> 3, cq = (t & 7) << 2;
#pragma unroll
    for (int p = 0; p < 4; ++p) {
        int r = (p << 5) + r0;
        float w0 = wr[p*4+0], w1 = wr[p*4+1], w2 = wr[p*4+2], w3 = wr[p*4+3];
        __bf16 h0 = (__bf16)w0, h1 = (__bf16)w1, h2 = (__bf16)w2, h3 = (__bf16)w3;
        v4bf hv = {h0, h1, h2, h3};
        v4bf lv = {(__bf16)(w0 - (float)h0), (__bf16)(w1 - (float)h1),
                   (__bf16)(w2 - (float)h2), (__bf16)(w3 - (float)h3)};
        *(v4bf*)&dBhi[r * LDSTRD + cq] = hv;
        *(v4bf*)&dBlo[r * LDSTRD + cq] = lv;
    }
}

// ---------------------------------------------------------------------------
// bf16x3 split-precision GEMM:  D[m,n] = sum_k A[m,k] * W[n,k]  (+bias)
// Double-buffered LDS; A staged via async-to-LDS (ASYNCcnt), W split on the fly.
// ---------------------------------------------------------------------------
template <bool FUSE_TANH>
__global__ __launch_bounds__(256) void gemm_bf16x3_kernel(
    const __bf16* __restrict__ Ahi, const __bf16* __restrict__ Alo, int aColOff,
    const float* __restrict__ W, int ldw, int N, int K, int Kreal,
    const float* __restrict__ bias,
    float* __restrict__ out, int ldo,
    __bf16* __restrict__ Ohi, __bf16* __restrict__ Olo)
{
    __shared__ __bf16 sAhi[2][128 * LDSTRD];
    __shared__ __bf16 sAlo[2][128 * LDSTRD];
    __shared__ __bf16 sBhi[2][128 * LDSTRD];
    __shared__ __bf16 sBlo[2][128 * LDSTRD];

    const int t    = threadIdx.x;
    const int lane = t & 31;
    const int wid  = t >> 5;
    const int wm   = wid >> 2;   // 0..1 -> 64-row patch
    const int wn   = wid & 3;    // 0..3 -> 32-col patch
    const int mBase = blockIdx.x * 128;
    const int nBase = blockIdx.y * 128;

    const v8f vzero = {0.f, 0.f, 0.f, 0.f, 0.f, 0.f, 0.f, 0.f};
    v8f acc[4][2];
#pragma unroll
    for (int i = 0; i < 4; ++i)
#pragma unroll
        for (int j = 0; j < 2; ++j) acc[i][j] = vzero;

    const int kTiles = K >> 5;
    float breg[16];

    // ---- prologue: fill buffer 0 with tile 0
    stage_A(Ahi, Alo, mBase, aColOff, 0, t, sAhi[0], sAlo[0]);
    load_B(W, ldw, N, Kreal, nBase, 0, t, breg);
    store_B(breg, t, sBhi[0], sBlo[0]);
    ASYNC_WAIT0();
    __syncthreads();

    for (int kt = 0; kt < kTiles; ++kt) {
        const int cur = kt & 1, nxt = cur ^ 1;
        const bool has_next = (kt + 1 < kTiles);

        // ---- issue next tile's global traffic; overlaps with WMMAs below.
        if (has_next) {
            const int kNext = (kt + 1) << 5;
            stage_A(Ahi, Alo, mBase, aColOff, kNext, t, sAhi[nxt], sAlo[nxt]);
            load_B(W, ldw, N, Kreal, nBase, kNext, t, breg);
        }

        // ---- compute on buffer `cur`: 24 WMMAs (bf16x3: hh + hl + lh)
        v16bf bh[2], bl[2];
#pragma unroll
        for (int j = 0; j < 2; ++j) {
            int n0 = (wn * 32 + j * 16) * LDSTRD;
            bh[j] = load_frag_b(&sBhi[cur][n0], lane);
            bl[j] = load_frag_b(&sBlo[cur][n0], lane);
        }
#pragma unroll
        for (int i = 0; i < 4; ++i) {
            int m0 = (wm * 64 + i * 16) * LDSTRD;
            v16bf ah = load_frag_a(&sAhi[cur][m0], lane);
            v16bf al = load_frag_a(&sAlo[cur][m0], lane);
#pragma unroll
            for (int j = 0; j < 2; ++j) {
                acc[i][j] = WMMA_BF16(ah, bh[j], acc[i][j]);
                acc[i][j] = WMMA_BF16(ah, bl[j], acc[i][j]);
                acc[i][j] = WMMA_BF16(al, bh[j], acc[i][j]);
            }
        }

        // ---- finish filling buffer `nxt`, hand off
        if (has_next) {
            store_B(breg, t, sBhi[nxt], sBlo[nxt]);
            ASYNC_WAIT0();
        }
        __syncthreads();
    }

    // ---- epilogue (C/D layout: lane n=lane&15, VGPR r -> m = r + 8*(lane>>4))
    const int nCol  = lane & 15;
    const int mHalf = (lane >> 4) * 8;
#pragma unroll
    for (int i = 0; i < 4; ++i)
#pragma unroll
        for (int j = 0; j < 2; ++j) {
            int n  = nBase + wn * 32 + j * 16 + nCol;
            int m0 = mBase + wm * 64 + i * 16 + mHalf;
            if (n < N) {
                float bv = bias[n];
#pragma unroll
                for (int r = 0; r < 8; ++r) {
                    float v = acc[i][j][r] + bv;
                    int m = m0 + r;
                    if constexpr (FUSE_TANH) {
                        float hvf = tanhf(v);
                        __bf16 hi = (__bf16)hvf;
                        __bf16 lo = (__bf16)(hvf - (float)hi);
                        size_t o = (size_t)m * XCOLS + n;
                        Ohi[o] = hi;
                        Olo[o] = lo;
                    } else {
                        // streaming result: keep it out of L2 (W2 + x want residency)
                        __builtin_nontemporal_store(v, &out[(size_t)m * ldo + n]);
                    }
                }
            }
        }
}

// ---------------------------------------------------------------------------
// Launch
// ---------------------------------------------------------------------------
extern "C" void kernel_launch(void* const* d_in, const int* in_sizes, int n_in,
                              void* d_out, int out_size, void* d_ws, size_t ws_size,
                              hipStream_t stream) {
    (void)in_sizes; (void)n_in; (void)out_size; (void)ws_size;
    const int*   ctx = (const int*)d_in[0];
    const float* We  = (const float*)d_in[1];
    const float* be  = (const float*)d_in[2];
    const float* W1  = (const float*)d_in[3];
    const float* b1  = (const float*)d_in[4];
    const float* W2  = (const float*)d_in[5];
    const float* b2  = (const float*)d_in[6];
    float* out = (float*)d_out;

    __bf16* xhi = (__bf16*)d_ws;                                      // [4096, 2048] bf16
    __bf16* xlo = (__bf16*)((char*)d_ws +
                            (size_t)BATCHN * XCOLS * sizeof(__bf16)); // [4096, 2048] bf16

    // 1) embedding gather + split -> x[:, 1000..2047]
    embed_split_kernel<<<BATCHN, 256, 0, stream>>>(ctx, We, be, xhi, xlo);

    // 2) h = tanh(embed @ W1^T + b1) -> split into x[:, 0..999]
    gemm_bf16x3_kernel<true><<<dim3(BATCHN / 128, (HIDN + 127) / 128), 256, 0, stream>>>(
        xhi, xlo, /*aColOff=*/HIDN,
        W1, /*ldw=*/CTXN * EMBN, /*N=*/HIDN, /*K=*/CTXN * EMBN, /*Kreal=*/CTXN * EMBN,
        b1, nullptr, 0, xhi, xlo);

    // 3) out = x @ W2^T + b2   (M-fastest grid: 32 M-blocks share each W2 tile via L2)
    gemm_bf16x3_kernel<false><<<dim3(BATCHN / 128, (VOCABN + 127) / 128), 256, 0, stream>>>(
        xhi, xlo, /*aColOff=*/0,
        W2, /*ldw=*/KREAL2, /*N=*/VOCABN, /*K=*/XCOLS, /*Kreal=*/KREAL2,
        b2, out, VOCABN, nullptr, nullptr);
}
